// DenseGATLayer_77206332113811
// MI455X (gfx1250) — matrix operand order repaired and checked by reading, not compiled, not run on hardware
//
#include <hip/hip_runtime.h>
#include <hip/hip_bf16.h>

typedef __bf16 bf16_t;
typedef __attribute__((ext_vector_type(16))) __bf16 v16bf;
typedef __attribute__((ext_vector_type(8)))  __bf16 v8bf;
typedef __attribute__((ext_vector_type(8)))  float  v8f;

union Frag {
    v16bf v;
    struct { v8bf lo; v8bf hi; } h;
};

#define WMMA_BF16(a, b, c) \
    __builtin_amdgcn_wmma_f32_16x16x32_bf16(false, (a), false, (b), (short)0, (c), false, false)

#if defined(__gfx1250__)
#  define USE_ASYNC_LDS 1
#else
#  define USE_ASYNC_LDS 0
#endif

// Async global->LDS copy of 16 bytes. vdst = wave-relative LDS byte offset
// (low 32 bits of the generic LDS address per the flat-aperture mapping),
// vaddr = 64-bit global address. Tracked by ASYNCcnt.
__device__ __forceinline__ void async_copy16(const void* g, void* l) {
#if USE_ASYNC_LDS
    const unsigned lds_off = (unsigned)(uintptr_t)l;
    asm volatile("global_load_async_to_lds_b128 %0, %1, off"
                 :
                 : "v"(lds_off), "v"(g)
                 : "memory");
#else
    *(v8bf*)l = *(const v8bf*)g;
#endif
}

__device__ __forceinline__ void wait_async_zero() {
#if USE_ASYNC_LDS
    asm volatile("s_wait_asynccnt 0x0" ::: "memory");
#endif
}

// ---------------------------------------------------------------------------
// Transpose + convert: w[K,N] f32 -> wt[N,K] bf16 (32x32 tiles via LDS)
// ---------------------------------------------------------------------------
__global__ __launch_bounds__(256) void transpose_bf16_kernel(const float* __restrict__ w,
                                                             bf16_t* __restrict__ wt,
                                                             int K, int N) {
    __shared__ float tile[32][33];
    const int k0  = blockIdx.y * 32;
    const int n0  = blockIdx.x * 32;
    const int tid = threadIdx.x;
    const int r   = tid >> 3;        // 0..31
    const int c   = (tid & 7) * 4;   // 0..28
#pragma unroll
    for (int j = 0; j < 4; ++j)
        tile[r][c + j] = w[(size_t)(k0 + r) * N + n0 + c + j];
    __syncthreads();
#pragma unroll
    for (int j = 0; j < 4; ++j)
        wt[(size_t)(n0 + r) * K + k0 + c + j] = (bf16_t)tile[c + j][r];
}

// ---------------------------------------------------------------------------
// LayerNorm over rows of 768, output bf16
// ---------------------------------------------------------------------------
__global__ __launch_bounds__(256) void layernorm_kernel(const float* __restrict__ xin,
                                                        const float* __restrict__ gam,
                                                        const float* __restrict__ bet,
                                                        bf16_t* __restrict__ out) {
    const int row = blockIdx.x;
    const int tid = threadIdx.x;
    const float* xr = xin + (size_t)row * 768;
    float v0 = xr[tid], v1 = xr[tid + 256], v2 = xr[tid + 512];

    __shared__ float red[16];
    float s = v0 + v1 + v2;
#pragma unroll
    for (int off = 16; off >= 1; off >>= 1) s += __shfl_xor(s, off, 32);
    const int w = tid >> 5, lane = tid & 31;
    if (lane == 0) red[w] = s;
    __syncthreads();
    float tot = 0.f;
#pragma unroll
    for (int i = 0; i < 8; ++i) tot += red[i];
    const float mean = tot * (1.0f / 768.0f);

    float d0 = v0 - mean, d1 = v1 - mean, d2 = v2 - mean;
    float s2 = d0 * d0 + d1 * d1 + d2 * d2;
#pragma unroll
    for (int off = 16; off >= 1; off >>= 1) s2 += __shfl_xor(s2, off, 32);
    if (lane == 0) red[8 + w] = s2;
    __syncthreads();
    float tot2 = 0.f;
#pragma unroll
    for (int i = 0; i < 8; ++i) tot2 += red[8 + i];
    const float rstd = rsqrtf(tot2 * (1.0f / 768.0f) + 1e-5f);

    bf16_t* orow = out + (size_t)row * 768;
    orow[tid]       = (bf16_t)(d0 * rstd * gam[tid]       + bet[tid]);
    orow[tid + 256] = (bf16_t)(d1 * rstd * gam[tid + 256] + bet[tid + 256]);
    orow[tid + 512] = (bf16_t)(d2 * rstd * gam[tid + 512] + bet[tid + 512]);
}

// ---------------------------------------------------------------------------
// Tiled bf16 GEMM with WMMA: C[M,N] = A[M,K] * Bt[N,K]^T + bias (+epilogue)
// Double-buffered LDS staging via async-to-LDS copies (fallback: b128 ld/st).
// Workgroup tile 128x64, K-tile 64 (8 WMMAs per stage), 8 waves @ 32x32.
// MODE 0: out bf16 = acc+bias
// MODE 1: out f32  = acc+bias+resid
// MODE 2: out bf16 = gelu(acc+bias)   (exact erf)
// Requires: M%128==0, N%64==0, K%64==0.
// ---------------------------------------------------------------------------
template <int MODE>
__global__ __launch_bounds__(256) void gemm_kernel(const bf16_t* __restrict__ A,
                                                   const bf16_t* __restrict__ Bt,
                                                   const float* __restrict__ bias,
                                                   const float* __restrict__ resid,
                                                   void* __restrict__ outp,
                                                   int M, int Kd, int Nd) {
    __shared__ bf16_t As[2][128 * 72];   // [row][k], stride 72 elem = 144 B
    __shared__ bf16_t Bs[2][64 * 72];    // [n][k]

    const int tid  = threadIdx.x;
    const int m0   = blockIdx.y * 128;
    const int n0   = blockIdx.x * 64;
    const int w    = tid >> 5, lane = tid & 31;
    const int half = lane >> 4, lrow = lane & 15;
    const int wm   = (w & 3) * 32;
    const int wn   = (w >> 2) * 32;
    const int ar   = tid >> 1, ac = (tid & 1) * 32;   // A: 128 rows x 64 k
    const int br   = tid >> 2, bc = (tid & 3) * 16;   // B: 64 rows x 64 k

    v8f c00 = {}, c01 = {}, c10 = {}, c11 = {};

    const bf16_t* aptr = A  + (size_t)(m0 + ar) * Kd + ac;
    const bf16_t* bptr = Bt + (size_t)(n0 + br) * Kd + bc;

    auto stage = [&](int k0, int buf) {
        bf16_t* ad = &As[buf][ar * 72 + ac];
        bf16_t* bd = &Bs[buf][br * 72 + bc];
        async_copy16(aptr + k0,      ad);
        async_copy16(aptr + k0 + 8,  ad + 8);
        async_copy16(aptr + k0 + 16, ad + 16);
        async_copy16(aptr + k0 + 24, ad + 24);
        async_copy16(bptr + k0,      bd);
        async_copy16(bptr + k0 + 8,  bd + 8);
    };

    const int nk = Kd >> 6;
    stage(0, 0);
    wait_async_zero();
    __syncthreads();

    for (int t = 0; t < nk; ++t) {
        const int cur = t & 1;
        if (t + 1 < nk) stage((t + 1) << 6, cur ^ 1);

        const bf16_t* Ab = As[cur];
        const bf16_t* Bb = Bs[cur];
#pragma unroll
        for (int kk = 0; kk < 64; kk += 32) {
            Frag fa0, fa1, fb0, fb1;
            fa0.h.lo = *(const v8bf*)(Ab + (wm + lrow) * 72 + kk + half * 8);
            fa0.h.hi = *(const v8bf*)(Ab + (wm + lrow) * 72 + kk + 16 + half * 8);
            fa1.h.lo = *(const v8bf*)(Ab + (wm + 16 + lrow) * 72 + kk + half * 8);
            fa1.h.hi = *(const v8bf*)(Ab + (wm + 16 + lrow) * 72 + kk + 16 + half * 8);
            fb0.h.lo = *(const v8bf*)(Bb + (wn + lrow) * 72 + kk + half * 8);
            fb0.h.hi = *(const v8bf*)(Bb + (wn + lrow) * 72 + kk + 16 + half * 8);
            fb1.h.lo = *(const v8bf*)(Bb + (wn + 16 + lrow) * 72 + kk + half * 8);
            fb1.h.hi = *(const v8bf*)(Bb + (wn + 16 + lrow) * 72 + kk + 16 + half * 8);

            c00 = WMMA_BF16(fa0.v, fb0.v, c00);
            c01 = WMMA_BF16(fa0.v, fb1.v, c01);
            c10 = WMMA_BF16(fa1.v, fb0.v, c10);
            c11 = WMMA_BF16(fa1.v, fb1.v, c11);
        }
        wait_async_zero();   // next buffer landed; reads of cur are done
        __syncthreads();
    }

    auto store_tile = [&](v8f c, int mt, int nt) {
#pragma unroll
        for (int j = 0; j < 8; ++j) {
            int gr = m0 + wm + mt * 16 + half * 8 + j;
            int gc = n0 + wn + nt * 16 + lrow;
            float r = c[j] + bias[gc];
            size_t idx = (size_t)gr * Nd + gc;
            if (MODE == 2) r = 0.5f * r * (1.0f + erff(r * 0.70710678118f));
            if (MODE == 1) {
                ((float*)outp)[idx] = r + resid[idx];
            } else {
                ((bf16_t*)outp)[idx] = (bf16_t)r;
            }
        }
    };
    store_tile(c00, 0, 0);
    store_tile(c01, 0, 1);
    store_tile(c10, 1, 0);
    store_tile(c11, 1, 1);
}

// ---------------------------------------------------------------------------
// Fused masked attention for one (b, h, 16-query tile).
// Q,K,V bf16 [B*512, 768] with head slice at col h*64. adj int32 [B,512,512].
// Output: attnout bf16 [B*512, 768].
// ---------------------------------------------------------------------------
__global__ __launch_bounds__(256) void attn_kernel(const bf16_t* __restrict__ Q,
                                                   const bf16_t* __restrict__ Km,
                                                   const bf16_t* __restrict__ V,
                                                   const int* __restrict__ adj,
                                                   bf16_t* __restrict__ Oout) {
    const int qt = blockIdx.x;   // 0..31 (16 query rows each)
    const int h  = blockIdx.y;   // 0..11
    const int b  = blockIdx.z;   // 0..31
    const int q0 = qt * 16;
    const int tid  = threadIdx.x;
    const int w    = tid >> 5, lane = tid & 31;
    const int half = lane >> 4, lrow = lane & 15;

    // 49,920 bytes total; P | (S aliased with Vt+Opart across barrier)
    __shared__ float smem[12480];
    bf16_t* Pb = (bf16_t*)smem;                 // [16][520] bf16  (bytes 0..16639)
    float*  S  = smem + 4160;                   // [16][520] f32   (bytes 16640..49919)
    bf16_t* Vt = (bf16_t*)(smem + 4160);        // [2][64][72] bf16 (aliases S, phase 3)
    float*  Op = smem + 4160 + 4608;            // [2][16][72] f32

    // ----- Phase 1: S[16][512] = Q_tile * K^T ---------------------------------
    const bf16_t* qrow = Q + ((size_t)(b * 512 + q0 + lrow)) * 768 + h * 64;
    Frag aq0, aq1;
    aq0.h.lo = *(const v8bf*)(qrow + half * 8);
    aq0.h.hi = *(const v8bf*)(qrow + 16 + half * 8);
    aq1.h.lo = *(const v8bf*)(qrow + 32 + half * 8);
    aq1.h.hi = *(const v8bf*)(qrow + 48 + half * 8);

    const bf16_t* Kbase = Km + ((size_t)(b * 512)) * 768 + h * 64;
#pragma unroll
    for (int nt = 0; nt < 4; ++nt) {
        const int kb = w * 64 + nt * 16;  // key-row base for this 16x16 tile
        const bf16_t* krow = Kbase + (size_t)(kb + lrow) * 768;
        Frag b0, b1;
        b0.h.lo = *(const v8bf*)(krow + half * 8);
        b0.h.hi = *(const v8bf*)(krow + 16 + half * 8);
        b1.h.lo = *(const v8bf*)(krow + 32 + half * 8);
        b1.h.hi = *(const v8bf*)(krow + 48 + half * 8);
        v8f c = {};
        c = WMMA_BF16(aq0.v, b0.v, c);
        c = WMMA_BF16(aq1.v, b1.v, c);
#pragma unroll
        for (int j = 0; j < 8; ++j) S[(half * 8 + j) * 520 + kb + lrow] = c[j];
    }
    __syncthreads();

    // ----- Phase 2: masked softmax per row (wave handles 2 rows) --------------
#pragma unroll
    for (int rr = 0; rr < 2; ++rr) {
        const int r = w * 2 + rr;
        const int* arow = adj + ((size_t)b * 512 + q0 + r) * 512;
        float vals[16];
        float m = -1e30f;
#pragma unroll
        for (int i = 0; i < 16; ++i) {
            const int c = lane + i * 32;
            float s = S[r * 520 + c] * 0.125f;  // scale = 64^-0.5
            vals[i] = (arow[c] != 0) ? s : -1e30f;
            m = fmaxf(m, vals[i]);
        }
#pragma unroll
        for (int off = 16; off >= 1; off >>= 1) m = fmaxf(m, __shfl_xor(m, off, 32));
        float sum = 0.f;
#pragma unroll
        for (int i = 0; i < 16; ++i) {
            vals[i] = expf(vals[i] - m);
            sum += vals[i];
        }
#pragma unroll
        for (int off = 16; off >= 1; off >>= 1) sum += __shfl_xor(sum, off, 32);
        const float inv = 1.0f / sum;  // self-loop guarantees sum > 0
#pragma unroll
        for (int i = 0; i < 16; ++i)
            Pb[r * 520 + lane + i * 32] = (bf16_t)(vals[i] * inv);
    }
    __syncthreads();  // last read of S; Vt may now overwrite it

    // ----- Phase 3: O[16][64] = P[16][512] * V[512][64], split-K over 2 groups -
    const int g = w >> 2, nt3 = w & 3;
    v8f oc = {};
    const bf16_t* Vbase = V + ((size_t)(b * 512)) * 768 + h * 64;
    for (int it = 0; it < 4; ++it) {
        // stage two 64x64 V tiles (one per split-K group), transposed to [hd][kv]
        const int gg  = tid >> 7;        // group staged by this thread
        const int tt  = tid & 127;
        const int kv  = tt >> 1;         // 0..63
        const int hd0 = (tt & 1) * 32;
        const bf16_t* vrow = Vbase + (size_t)(gg * 256 + it * 64 + kv) * 768 + hd0;
        v8bf v0 = ((const v8bf*)vrow)[0];
        v8bf v1 = ((const v8bf*)vrow)[1];
        v8bf v2 = ((const v8bf*)vrow)[2];
        v8bf v3 = ((const v8bf*)vrow)[3];
        __syncthreads();
        bf16_t* vt = Vt + gg * 64 * 72;
#pragma unroll
        for (int j = 0; j < 8; ++j) {
            vt[(hd0 + j) * 72 + kv]      = v0[j];
            vt[(hd0 + 8 + j) * 72 + kv]  = v1[j];
            vt[(hd0 + 16 + j) * 72 + kv] = v2[j];
            vt[(hd0 + 24 + j) * 72 + kv] = v3[j];
        }
        __syncthreads();
#pragma unroll
        for (int ks = 0; ks < 2; ++ks) {
            const int kbase = g * 256 + it * 64 + ks * 32;  // kv base for A frag
            Frag ap, bv;
            const bf16_t* prow = Pb + lrow * 520 + kbase;
            ap.h.lo = *(const v8bf*)(prow + half * 8);
            ap.h.hi = *(const v8bf*)(prow + 16 + half * 8);
            const bf16_t* vtrow = Vt + g * 64 * 72 + (nt3 * 16 + lrow) * 72 + ks * 32;
            bv.h.lo = *(const v8bf*)(vtrow + half * 8);
            bv.h.hi = *(const v8bf*)(vtrow + 16 + half * 8);
            oc = WMMA_BF16(ap.v, bv.v, oc);
        }
    }
    // partial results to LDS, then combine the two split-K groups
#pragma unroll
    for (int j = 0; j < 8; ++j)
        Op[(g * 16 + half * 8 + j) * 72 + nt3 * 16 + lrow] = oc[j];
    __syncthreads();
    {
        const int r  = tid >> 4;         // 0..15
        const int c0 = (tid & 15) * 4;   // 0..60
        bf16_t* orow = Oout + ((size_t)(b * 512 + q0 + r)) * 768 + h * 64 + c0;
#pragma unroll
        for (int j = 0; j < 4; ++j) {
            float o = Op[r * 72 + c0 + j] + Op[16 * 72 + r * 72 + c0 + j];
            orow[j] = (bf16_t)o;
        }
    }
}

// ---------------------------------------------------------------------------
// Host-side orchestration
// ---------------------------------------------------------------------------
extern "C" void kernel_launch(void* const* d_in, const int* in_sizes, int n_in,
                              void* d_out, int out_size, void* d_ws, size_t ws_size,
                              hipStream_t stream) {
    (void)in_sizes; (void)n_in; (void)out_size; (void)ws_size;
    const int M = 32 * 512;          // 16384 rows
    const int D = 768, D4 = 3072;

    const float* x      = (const float*)d_in[0];
    const int*   adj    = (const int*)d_in[1];
    const float* wq     = (const float*)d_in[2];
    const float* bq     = (const float*)d_in[3];
    const float* wk     = (const float*)d_in[4];
    const float* bk     = (const float*)d_in[5];
    const float* wv     = (const float*)d_in[6];
    const float* bv     = (const float*)d_in[7];
    const float* wo     = (const float*)d_in[8];
    const float* bo     = (const float*)d_in[9];
    const float* g1     = (const float*)d_in[10];
    const float* b1     = (const float*)d_in[11];
    const float* g2     = (const float*)d_in[12];
    const float* b2     = (const float*)d_in[13];
    const float* w_ffn1 = (const float*)d_in[14];
    const float* b_ffn1 = (const float*)d_in[15];
    const float* w_ffn2 = (const float*)d_in[16];
    const float* b_ffn2 = (const float*)d_in[17];

    char* ws = (char*)d_ws;
    size_t off = 0;
    auto alloc = [&](size_t bytes) -> char* {
        char* p = ws + off;
        off += (bytes + 255) & ~(size_t)255;
        return p;
    };
    const size_t szDD  = (size_t)D * D;     // 589824
    const size_t szDD4 = (size_t)D * D4;    // 2359296
    const size_t szAct = (size_t)M * D;     // 12582912 elements

    // pre-transposed bf16 weights: wt[N][K]
    bf16_t* wq_t = (bf16_t*)alloc(szDD * 2);
    bf16_t* wk_t = (bf16_t*)alloc(szDD * 2);
    bf16_t* wv_t = (bf16_t*)alloc(szDD * 2);
    bf16_t* wo_t = (bf16_t*)alloc(szDD * 2);
    bf16_t* w1_t = (bf16_t*)alloc(szDD4 * 2);
    bf16_t* w2_t = (bf16_t*)alloc(szDD4 * 2);
    bf16_t* h_b  = (bf16_t*)alloc(szAct * 2);   // reused for h2
    bf16_t* q_b  = (bf16_t*)alloc(szAct * 2);   // q,k,v,a contiguous: f1 aliases them
    bf16_t* k_b  = (bf16_t*)alloc(szAct * 2);
    bf16_t* v_b  = (bf16_t*)alloc(szAct * 2);
    bf16_t* a_b  = (bf16_t*)alloc(szAct * 2);
    float*  out_f = (float*)alloc(szAct * 4);
    bf16_t* f1_b = q_b;                         // [M, 3072] aliases q/k/v/a

    dim3 blk(256);
    // weight transpose+convert (w[K,N] -> wt[N,K])
    transpose_bf16_kernel<<<dim3(D / 32, D / 32),  blk, 0, stream>>>(wq, wq_t, D, D);
    transpose_bf16_kernel<<<dim3(D / 32, D / 32),  blk, 0, stream>>>(wk, wk_t, D, D);
    transpose_bf16_kernel<<<dim3(D / 32, D / 32),  blk, 0, stream>>>(wv, wv_t, D, D);
    transpose_bf16_kernel<<<dim3(D / 32, D / 32),  blk, 0, stream>>>(wo, wo_t, D, D);
    transpose_bf16_kernel<<<dim3(D4 / 32, D / 32), blk, 0, stream>>>(w_ffn1, w1_t, D, D4);
    transpose_bf16_kernel<<<dim3(D / 32, D4 / 32), blk, 0, stream>>>(w_ffn2, w2_t, D4, D);

    // LN1: h = LN(x, g1, b1)
    layernorm_kernel<<<M, 256, 0, stream>>>(x, g1, b1, h_b);

    dim3 g768(D / 64, M / 128);    // (12, 128)
    dim3 g3072(D4 / 64, M / 128);  // (48, 128)

    // Q, K, V projections
    gemm_kernel<0><<<g768, blk, 0, stream>>>(h_b, wq_t, bq, nullptr, q_b, M, D, D);
    gemm_kernel<0><<<g768, blk, 0, stream>>>(h_b, wk_t, bk, nullptr, k_b, M, D, D);
    gemm_kernel<0><<<g768, blk, 0, stream>>>(h_b, wv_t, bv, nullptr, v_b, M, D, D);

    // masked attention
    attn_kernel<<<dim3(32, 12, 32), blk, 0, stream>>>(q_b, k_b, v_b, adj, a_b);

    // out = attnout @ wo + bo + x   (f32)
    gemm_kernel<1><<<g768, blk, 0, stream>>>(a_b, wo_t, bo, x, out_f, M, D, D);

    // h2 = LN(out, g2, b2)
    layernorm_kernel<<<M, 256, 0, stream>>>(out_f, g2, b2, h_b);

    // f1 = gelu(h2 @ w1 + b1)
    gemm_kernel<2><<<g3072, blk, 0, stream>>>(h_b, w1_t, b_ffn1, nullptr, f1_b, M, D, D4);

    // result = f1 @ w2 + b2 + out
    gemm_kernel<1><<<g768, blk, 0, stream>>>(f1_b, w2_t, b_ffn2, out_f, (float*)d_out, M, D4, D);
}
